// Q_ResBlockBn_unfold_71236327572143
// MI455X (gfx1250) — 8-bit matrix operand order repaired and checked by reading, compiled, not run on hardware
//
#include <hip/hip_runtime.h>
#include <hip/hip_bf16.h>

typedef __attribute__((ext_vector_type(8))) int v8i;
typedef __attribute__((ext_vector_type(4))) int v4i;

#define NIMG   32
#define CIN    128
#define COUT   256
#define HIN    56
#define WIN    56
#define HOUT   28
#define WOUT   28
#define MTOT   (NIMG * HOUT * WOUT)
#define K1     (CIN * 9)
#define K2     (COUT * 9)
#define KID    (CIN)
#define KB1    (K1 / 64)
#define KB2    (K2 / 64)
#define KBID   (KID / 64)
#define NELEM  (COUT * MTOT)

static const size_t OFF_A1   = 16384;
static const size_t SZ_A1    = (size_t)MTOT * K1;
static const size_t OFF_AID  = OFF_A1 + SZ_A1;
static const size_t SZ_AID   = (size_t)MTOT * KID;
static const size_t OFF_A2   = OFF_AID + SZ_AID;
static const size_t SZ_A2    = (size_t)MTOT * K2;
static const size_t OFF_B1   = OFF_A2 + SZ_A2;
static const size_t OFF_B2   = OFF_B1 + (size_t)K1 * COUT;
static const size_t OFF_BID  = OFF_B2 + (size_t)K2 * COUT;
static const size_t OFF_ACC1 = OFF_BID + (size_t)KID * COUT;
static const size_t SZ_ACC   = (size_t)COUT * MTOT * 4;
static const size_t OFF_ACCID= OFF_ACC1 + SZ_ACC;
static const size_t OFF_HBN  = OFF_ACCID + SZ_ACC;
static const size_t OFF_SUM  = OFF_A1;


__device__ __forceinline__ float qclip(float v, float s) {
    float q = rintf(v / s);
    q = fminf(fmaxf(q, -128.f), 127.f);
    return q * s;
}
__device__ __forceinline__ float waveMaxF(float v) {
    #pragma unroll
    for (int o = 16; o > 0; o >>= 1) v = fmaxf(v, __shfl_xor(v, o, 32));
    return v;
}
__device__ __forceinline__ unsigned waveMaxU(unsigned v) {
    #pragma unroll
    for (int o = 16; o > 0; o >>= 1) { unsigned w = __shfl_xor(v, o, 32); v = v > w ? v : w; }
    return v;
}

__global__ void init_scalars_kernel(unsigned* wsu) {
    if (threadIdx.x < 64) wsu[threadIdx.x] = 0u;
}

__global__ void absmax_kernel(const float* __restrict__ p, int n, unsigned* slot) {
    float lm = 0.f;
    for (int i = blockIdx.x * blockDim.x + threadIdx.x; i < n; i += gridDim.x * blockDim.x)
        lm = fmaxf(lm, fabsf(p[i]));
    lm = waveMaxF(lm);
    if ((threadIdx.x & 31) == 0) atomicMax(slot, __float_as_uint(lm));
}

__global__ void finalize_kernel(float* wsf, int stage) {
    unsigned* u = (unsigned*)wsf;
    if (stage == 0) {
        wsf[4] = fmaxf(__uint_as_float(u[0]) / 127.f, 1e-8f);
        wsf[5] = fmaxf(__uint_as_float(u[1]) / 127.f, 1e-8f);
        wsf[6] = fmaxf(__uint_as_float(u[2]) / 127.f, 1e-8f);
        wsf[7] = fmaxf(__uint_as_float(u[3]) / 127.f, 1e-8f);
    } else if (stage == 1) {
        wsf[11] = fmaxf(wsf[4] * wsf[5] * (float)u[8]  / 127.f, 1e-8f);
        wsf[14] = fmaxf(wsf[4] * wsf[7] * (float)u[10] / 127.f, 1e-8f);
    } else if (stage == 2) {
        wsf[12] = fmaxf(__uint_as_float(u[15]) / 127.f, 1e-8f);
    } else if (stage == 3) {
        wsf[13] = fmaxf(wsf[12] * wsf[6] * (float)u[9] / 127.f, 1e-8f);
    } else {
        wsf[17] = fmaxf(__uint_as_float(u[16]) / 127.f, 1e-8f);
    }
}

__global__ void bnprep_kernel(const float* g0, const float* be0, const float* me0, const float* va0,
                              const float* g1, const float* be1, const float* me1, const float* va1,
                              const float* g2, const float* be2, const float* me2, const float* va2,
                              float* wsf) {
    const float* G[3]  = { g0, g1, g2 };
    const float* BE[3] = { be0, be1, be2 };
    const float* ME[3] = { me0, me1, me2 };
    const float* VA[3] = { va0, va1, va2 };
    float* WQ[3] = { wsf + 64,  wsf + 576,  wsf + 1088 };
    float* BB[3] = { wsf + 320, wsf + 832,  wsf + 1344 };
    __shared__ float red[256];
    int c = threadIdx.x;
    for (int s = 0; s < 3; ++s) {
        float wsv = G[s][c] * rsqrtf(VA[s][c] + 1e-5f);
        red[c] = fabsf(wsv);
        __syncthreads();
        for (int o = 128; o > 0; o >>= 1) {
            if (c < o) red[c] = fmaxf(red[c], red[c + o]);
            __syncthreads();
        }
        float sw = fmaxf(red[0] / 127.f, 1e-8f);
        __syncthreads();
        float wq = fminf(fmaxf(rintf(wsv / sw), -128.f), 127.f) * sw;
        WQ[s][c] = wq;
        BB[s][c] = BE[s][c] - ME[s][c] * wq;
        __syncthreads();
    }
}

__global__ void packw_kernel(const float* __restrict__ w, unsigned char* __restrict__ dst,
                             const float* sSlot, int Cc, int KH, int KW, int kblks, int total) {
    int t = blockIdx.x * blockDim.x + threadIdx.x;
    if (t >= total) return;
    int K  = Cc * KH * KW;
    int kq = K >> 2;
    int o  = t / kq;
    int k4 = (t - o * kq) * 4;
    float s = *sSlot;
    unsigned pack = 0;
    #pragma unroll
    for (int j = 0; j < 4; ++j) {
        int k = k4 + j;
        int c = k % Cc, r = k / Cc;
        int kh = r / KW, kw_ = r % KW;
        float v = w[(((size_t)o * Cc + c) * KH + kh) * KW + kw_];
        int qi = (int)fminf(fmaxf(rintf(v / s), -128.f), 127.f);
        pack |= ((unsigned)(qi & 0xFF)) << (8 * j);
    }
    int kk = k4 & 63, kb = k4 >> 6;
    int nblk = o >> 4;
    int lane = (o & 15) + ((kk >= 32) ? 16 : 0);
    int b = kk & 31;
    *(unsigned*)(dst + ((((size_t)nblk * kblks + kb) * 32 + lane) * 32 + b)) = pack;
}

__global__ void packa_kernel(const float* __restrict__ src, unsigned char* __restrict__ dst,
                             const float* sSlot,
                             int C, int H, int W, int OH, int OW, int KH, int KW,
                             int stride, int pad, int kblks,
                             long sn, long sc, long sh, int total) {
    int t = blockIdx.x * blockDim.x + threadIdx.x;
    if (t >= total) return;
    int K  = C * KH * KW;
    int kq = K >> 2;
    int m  = t / kq;
    int k4 = (t - m * kq) * 4;
    int nimg = m / (OH * OW);
    int rem  = m - nimg * (OH * OW);
    int oh = rem / OW, ow = rem - (rem / OW) * OW;
    float s = *sSlot;
    int c0 = k4 % C, r = k4 / C;
    int kh = r / KW, kw_ = r % KW;
    int ih = oh * stride - pad + kh;
    int iw = ow * stride - pad + kw_;
    bool inb = (ih >= 0) && (ih < H) && (iw >= 0) && (iw < W);
    unsigned pack = 0;
    #pragma unroll
    for (int j = 0; j < 4; ++j) {
        float v = inb ? src[(size_t)nimg * sn + (size_t)(c0 + j) * sc + (size_t)ih * sh + iw] : 0.f;
        int qi = (int)fminf(fmaxf(rintf(v / s), -128.f), 127.f);
        pack |= ((unsigned)(qi & 0xFF)) << (8 * j);
    }
    int kk = k4 & 63, kb = k4 >> 6;
    int mblk = m >> 4;
    int lane = (m & 15) + ((kk >= 32) ? 16 : 0);
    int b = kk & 31;
    *(unsigned*)(dst + ((((size_t)mblk * kblks + kb) * 32 + lane) * 32 + b)) = pack;
}

template <int KBLKS>
__global__ __launch_bounds__(256)
void gemm_iu8_kernel(const unsigned char* __restrict__ Ap,
                     const unsigned char* __restrict__ Bp,
                     int* __restrict__ acc, unsigned* __restrict__ imaxSlot) {
    const int M = MTOT;
    int lane = threadIdx.x & 31;
    int wave = threadIdx.x >> 5;
    int mi = wave & 1, ni = wave >> 1;
    int mbase = blockIdx.x * 64 + mi * 32;
    int nbase = ni * 64;
    int mblk = mbase >> 4;
    int nblk = nbase >> 4;

    const v8i* a0 = (const v8i*)Ap + ((size_t)(mblk + 0) * KBLKS * 32 + lane);
    const v8i* a1 = (const v8i*)Ap + ((size_t)(mblk + 1) * KBLKS * 32 + lane);
    const v8i* b0 = (const v8i*)Bp + ((size_t)(nblk + 0) * KBLKS * 32 + lane);
    const v8i* b1 = (const v8i*)Bp + ((size_t)(nblk + 1) * KBLKS * 32 + lane);
    const v8i* b2 = (const v8i*)Bp + ((size_t)(nblk + 2) * KBLKS * 32 + lane);
    const v8i* b3 = (const v8i*)Bp + ((size_t)(nblk + 3) * KBLKS * 32 + lane);

    v8i z = {0, 0, 0, 0, 0, 0, 0, 0};
    v8i cfr[2][4];
    #pragma unroll
    for (int i = 0; i < 2; ++i)
        #pragma unroll
        for (int j = 0; j < 4; ++j) cfr[i][j] = z;

    #pragma unroll
    for (int kb = 0; kb < KBLKS; ++kb) {
        v8i A0 = a0[kb * 32];
        v8i A1 = a1[kb * 32];
        v8i B0 = b0[kb * 32];
        v8i B1 = b1[kb * 32];
        v8i B2 = b2[kb * 32];
        v8i B3 = b3[kb * 32];
        cfr[0][0] = __builtin_amdgcn_wmma_i32_16x16x64_iu8(true, A0, true, B0, cfr[0][0], false, false);
        cfr[0][1] = __builtin_amdgcn_wmma_i32_16x16x64_iu8(true, A0, true, B1, cfr[0][1], false, false);
        cfr[0][2] = __builtin_amdgcn_wmma_i32_16x16x64_iu8(true, A0, true, B2, cfr[0][2], false, false);
        cfr[0][3] = __builtin_amdgcn_wmma_i32_16x16x64_iu8(true, A0, true, B3, cfr[0][3], false, false);
        cfr[1][0] = __builtin_amdgcn_wmma_i32_16x16x64_iu8(true, A1, true, B0, cfr[1][0], false, false);
        cfr[1][1] = __builtin_amdgcn_wmma_i32_16x16x64_iu8(true, A1, true, B1, cfr[1][1], false, false);
        cfr[1][2] = __builtin_amdgcn_wmma_i32_16x16x64_iu8(true, A1, true, B2, cfr[1][2], false, false);
        cfr[1][3] = __builtin_amdgcn_wmma_i32_16x16x64_iu8(true, A1, true, B3, cfr[1][3], false, false);
    }

    unsigned lm = 0;
    #pragma unroll
    for (int i = 0; i < 2; ++i) {
        #pragma unroll
        for (int j = 0; j < 4; ++j) {
            int n = nbase + j * 16 + (lane & 15);
            size_t base = (size_t)n * M + (size_t)(mbase + i * 16 + 8 * (lane >> 4));
            v4i lo = { cfr[i][j][0], cfr[i][j][1], cfr[i][j][2], cfr[i][j][3] };
            v4i hi = { cfr[i][j][4], cfr[i][j][5], cfr[i][j][6], cfr[i][j][7] };
            *(v4i*)(acc + base)     = lo;
            *(v4i*)(acc + base + 4) = hi;
            #pragma unroll
            for (int v = 0; v < 8; ++v) {
                int val = cfr[i][j][v];
                unsigned av = (unsigned)(val < 0 ? -val : val);
                lm = lm > av ? lm : av;
            }
        }
    }
    lm = waveMaxU(lm);
    if (lane == 0) atomicMax(imaxSlot, lm);
}

__global__ void ew1_kernel(const int* __restrict__ acc, const float* wsp,
                           float* __restrict__ hbn, unsigned* fmaxSlot, int n) {
    float sH = wsp[4] * wsp[5];
    float s1 = wsp[11];
    float lm = 0.f;
    for (int i = blockIdx.x * blockDim.x + threadIdx.x; i < n; i += gridDim.x * blockDim.x) {
        int c = i / MTOT;
        float h  = (float)acc[i] * sH;
        float hq = qclip(h, s1);
        float hb = hq * wsp[64 + c] + wsp[320 + c];
        float hr = fmaxf(hb, 0.f);
        hbn[i] = hr;
        lm = fmaxf(lm, hr);
    }
    lm = waveMaxF(lm);
    if ((threadIdx.x & 31) == 0) atomicMax(fmaxSlot, __float_as_uint(lm));
}

__global__ void ew2_kernel(const int* __restrict__ acc2, const int* __restrict__ accid,
                           const float* wsp, float* __restrict__ sum,
                           unsigned* fmaxSlot, int n) {
    float sH2  = wsp[12] * wsp[6];
    float s3   = wsp[13];
    float sHid = wsp[4] * wsp[7];
    float sid  = wsp[14];
    float lm = 0.f;
    for (int i = blockIdx.x * blockDim.x + threadIdx.x; i < n; i += gridDim.x * blockDim.x) {
        int c = i / MTOT;
        float h  = qclip((float)acc2[i]  * sH2,  s3) * wsp[576 + c]  + wsp[832 + c];
        float id = qclip((float)accid[i] * sHid, sid) * wsp[1088 + c] + wsp[1344 + c];
        float sv = h + id;
        sum[i] = sv;
        lm = fmaxf(lm, fabsf(sv));
    }
    lm = waveMaxF(lm);
    if ((threadIdx.x & 31) == 0) atomicMax(fmaxSlot, __float_as_uint(lm));
}

__global__ void final_kernel(const float* __restrict__ sum, const float* wsp,
                             float* __restrict__ out, int n) {
    float so = wsp[17];
    for (int i = blockIdx.x * blockDim.x + threadIdx.x; i < n; i += gridDim.x * blockDim.x) {
        int c = i / MTOT;
        int m = i - c * MTOT;
        int nimg = m / (HOUT * WOUT);
        int p = m - nimg * (HOUT * WOUT);
        float v = fmaxf(qclip(sum[i], so), 0.f);
        out[((size_t)nimg * COUT + c) * (HOUT * WOUT) + p] = v;
    }
    if (blockIdx.x == 0 && threadIdx.x == 0) out[NELEM] = so;
}

extern "C" void kernel_launch(void* const* d_in, const int* in_sizes, int n_in,
                              void* d_out, int out_size, void* d_ws, size_t ws_size,
                              hipStream_t stream) {
    const float* x   = (const float*)d_in[0];
    const float* w1  = (const float*)d_in[1];
    const float* w2  = (const float*)d_in[2];
    const float* wid = (const float*)d_in[3];
    const float* g1  = (const float*)d_in[4],  *be1 = (const float*)d_in[5];
    const float* me1 = (const float*)d_in[6],  *va1 = (const float*)d_in[7];
    const float* g2  = (const float*)d_in[8],  *be2 = (const float*)d_in[9];
    const float* me2 = (const float*)d_in[10], *va2 = (const float*)d_in[11];
    const float* gI  = (const float*)d_in[12], *beI = (const float*)d_in[13];
    const float* meI = (const float*)d_in[14], *vaI = (const float*)d_in[15];
    float* out = (float*)d_out;

    unsigned char* wsb = (unsigned char*)d_ws;
    float*    wsf = (float*)d_ws;
    unsigned* wsu = (unsigned*)d_ws;

    unsigned char* A1  = wsb + OFF_A1;
    unsigned char* AID = wsb + OFF_AID;
    unsigned char* A2  = wsb + OFF_A2;
    unsigned char* B1  = wsb + OFF_B1;
    unsigned char* B2  = wsb + OFF_B2;
    unsigned char* BID = wsb + OFF_BID;
    int*   acc1  = (int*)(wsb + OFF_ACC1);
    int*   accid = (int*)(wsb + OFF_ACCID);
    float* hbn   = (float*)(wsb + OFF_HBN);
    float* sum   = (float*)(wsb + OFF_SUM);

    init_scalars_kernel<<<1, 64, 0, stream>>>(wsu);
    absmax_kernel<<<1024, 256, 0, stream>>>(x,   NIMG * CIN * HIN * WIN, wsu + 0);
    absmax_kernel<<<128,  256, 0, stream>>>(w1,  COUT * CIN * 9,         wsu + 1);
    absmax_kernel<<<128,  256, 0, stream>>>(w2,  COUT * COUT * 9,        wsu + 2);
    absmax_kernel<<<16,   256, 0, stream>>>(wid, COUT * CIN,             wsu + 3);
    finalize_kernel<<<1, 1, 0, stream>>>(wsf, 0);
    bnprep_kernel<<<1, 256, 0, stream>>>(g1, be1, me1, va1, g2, be2, me2, va2, gI, beI, meI, vaI, wsf);

    packw_kernel<<<288, 256, 0, stream>>>(w1,  B1,  wsf + 5, CIN,  3, 3, KB1,  COUT * (K1 / 4));
    packw_kernel<<<576, 256, 0, stream>>>(w2,  B2,  wsf + 6, COUT, 3, 3, KB2,  COUT * (K2 / 4));
    packw_kernel<<<32,  256, 0, stream>>>(wid, BID, wsf + 7, CIN,  1, 1, KBID, COUT * (KID / 4));

    packa_kernel<<<28224, 256, 0, stream>>>(x, A1, wsf + 4, CIN, HIN, WIN, HOUT, WOUT,
                                            3, 3, 2, 1, KB1,
                                            (long)CIN * HIN * WIN, (long)HIN * WIN, (long)WIN,
                                            MTOT * (K1 / 4));
    packa_kernel<<<3136, 256, 0, stream>>>(x, AID, wsf + 4, CIN, HIN, WIN, HOUT, WOUT,
                                           1, 1, 2, 0, KBID,
                                           (long)CIN * HIN * WIN, (long)HIN * WIN, (long)WIN,
                                           MTOT * (KID / 4));

    gemm_iu8_kernel<KB1><<<MTOT / 64, 256, 0, stream>>>(A1,  B1,  acc1,  wsu + 8);
    gemm_iu8_kernel<KBID><<<MTOT / 64, 256, 0, stream>>>(AID, BID, accid, wsu + 10);
    finalize_kernel<<<1, 1, 0, stream>>>(wsf, 1);

    ew1_kernel<<<6272, 256, 0, stream>>>(acc1, wsf, hbn, wsu + 15, NELEM);
    finalize_kernel<<<1, 1, 0, stream>>>(wsf, 2);

    packa_kernel<<<56448, 256, 0, stream>>>(hbn, A2, wsf + 12, COUT, HOUT, WOUT, HOUT, WOUT,
                                            3, 3, 1, 1, KB2,
                                            (long)(HOUT * WOUT), (long)MTOT, (long)WOUT,
                                            MTOT * (K2 / 4));
    gemm_iu8_kernel<KB2><<<MTOT / 64, 256, 0, stream>>>(A2, B2, acc1, wsu + 9);
    finalize_kernel<<<1, 1, 0, stream>>>(wsf, 3);

    ew2_kernel<<<6272, 256, 0, stream>>>(acc1, accid, wsf, sum, wsu + 16, NELEM);
    finalize_kernel<<<1, 1, 0, stream>>>(wsf, 4);

    final_kernel<<<6272, 256, 0, stream>>>(sum, wsf, out, NELEM);
}
